// HetroGNN_1340029796804
// MI455X (gfx1250) — matrix operand order repaired and checked by reading, not compiled, not run on hardware
//
#include <hip/hip_runtime.h>
#include <hip/hip_bf16.h>

typedef __attribute__((ext_vector_type(16))) _Float16 v16h;
typedef __attribute__((ext_vector_type(8)))  float    v8f;

#define WMMA16(a, b, c) __builtin_amdgcn_wmma_f32_16x16x32_f16( \
    false, (a), false, (b), (short)0, (c), false, false)

#define GNN_N 50000
#define GNN_E 800000
#define NEGS 0.2f

// A-fragment column index for 16-bit 16x32 A (ISA 7.12.2):
// element i (0..15), hi = lane/16
__device__ __forceinline__ int a_col(int i, int hi) {
    int v = i >> 1;
    return ((v & 3) << 1) + ((v >> 2) << 4) + (hi << 3) + (i & 1);
}

__device__ __forceinline__ float lrelu(float x) { return x > 0.f ? x : NEGS * x; }

// ---------------------------------------------------------------------------
// Pack a (nMats x K x 64) row-major f32 weight into WMMA B fragments (f16).
// Output index: (((m*KC + kc)*4 + nt)*32 + lane)*16 + i
// B layout: lane = N (0..15) | hi=lane/16; element i -> K = hi*8 + (i&7) + (i>>3)*16
// ---------------------------------------------------------------------------
__global__ void pack_b_kernel(const float* __restrict__ W, _Float16* __restrict__ out,
                              int K, int nMats) {
    int KC = K >> 5;
    int total = nMats * KC * 4 * 32 * 16;
    int tid = blockIdx.x * blockDim.x + threadIdx.x;
    if (tid >= total) return;
    int i    = tid & 15;
    int lane = (tid >> 4) & 31;
    int nt   = (tid >> 9) & 3;
    int rest = tid >> 11;            // m*KC + kc
    int kc   = rest % KC;
    int m    = rest / KC;
    int k = kc * 32 + ((lane >> 4) << 3) + (i & 7) + ((i >> 3) << 4);
    int n = nt * 16 + (lane & 15);
    out[tid] = (_Float16)W[((size_t)m * K + k) * 64 + n];
}

// ---------------------------------------------------------------------------
// Node kernel: h = x @ Whl[type] + bhl[type]  (per-type WMMA + cndmask select),
// then mA = h @ WlinA  (h restaged f16 through LDS into A layout).
// One wave = 16 nodes; packed weights staged once per block into LDS.
// ---------------------------------------------------------------------------
__global__ __launch_bounds__(256) void node_hl_kernel(
    const float* __restrict__ xin, int relu_in,
    const int* __restrict__ ntype,
    const float* __restrict__ bhl,        // 3 x 64
    const _Float16* __restrict__ pWhl,    // 3 mats x 4096 halves
    const _Float16* __restrict__ pWlinA,  // 1 mat (4096 halves)
    float* __restrict__ hout,             // N x 64
    float* __restrict__ mAout,            // N x 64
    int nTiles) {
    __shared__ _Float16 wsh[16384];        // 32KB: [0,12288) Whl frags, [12288,16384) WlinA
    __shared__ _Float16 hstage[8][16][64]; // 16KB restage buffer
    // cooperative weight staging (b128 copies)
    {
        uint4* dsh = (uint4*)wsh;
        const uint4* s0 = (const uint4*)pWhl;    // 1536 x uint4
        for (int idx = threadIdx.x; idx < 1536; idx += 256) dsh[idx] = s0[idx];
        const uint4* s1 = (const uint4*)pWlinA;  // 512 x uint4
        for (int idx = threadIdx.x; idx < 512; idx += 256) dsh[1536 + idx] = s1[idx];
    }
    __syncthreads();

    int wave = threadIdx.x >> 5;
    int lane = threadIdx.x & 31;
    int tile = blockIdx.x * 8 + wave;
    bool active = tile < nTiles;
    int row0 = tile * 16;
    int hi   = lane >> 4;
    int mrow = row0 + (lane & 15);

    if (active) {
        // preload row types once (same rows for every n-tile)
        int trow[8];
#pragma unroll
        for (int r = 0; r < 8; ++r) trow[r] = ntype[row0 + r + hi * 8];

        v16h a0 = {}, a1 = {};
        const float* xrow = xin + (size_t)mrow * 64;
        __builtin_prefetch(xin + (size_t)(mrow + 128) * 64, 0, 1);
#pragma unroll
        for (int j = 0; j < 8; ++j) {
            int c0 = a_col(2 * j, hi);
            float2 p0 = *(const float2*)(xrow + c0);
            float2 p1 = *(const float2*)(xrow + 32 + c0);
            if (relu_in) {
                p0.x = fmaxf(p0.x, 0.f); p0.y = fmaxf(p0.y, 0.f);
                p1.x = fmaxf(p1.x, 0.f); p1.y = fmaxf(p1.y, 0.f);
            }
            a0[2 * j] = (_Float16)p0.x; a0[2 * j + 1] = (_Float16)p0.y;
            a1[2 * j] = (_Float16)p1.x; a1[2 * j + 1] = (_Float16)p1.y;
        }
#pragma unroll
        for (int nt = 0; nt < 4; ++nt) {
            int fo0 = ((0 * 4 + nt) * 32 + lane) * 16;
            int fo1 = ((1 * 4 + nt) * 32 + lane) * 16;
            v8f acc0 = {}, acc1 = {}, acc2 = {};
            acc0 = WMMA16(a0, *(const v16h*)(wsh + 0 * 4096 + fo0), acc0);
            acc0 = WMMA16(a1, *(const v16h*)(wsh + 0 * 4096 + fo1), acc0);
            acc1 = WMMA16(a0, *(const v16h*)(wsh + 1 * 4096 + fo0), acc1);
            acc1 = WMMA16(a1, *(const v16h*)(wsh + 1 * 4096 + fo1), acc1);
            acc2 = WMMA16(a0, *(const v16h*)(wsh + 2 * 4096 + fo0), acc2);
            acc2 = WMMA16(a1, *(const v16h*)(wsh + 2 * 4096 + fo1), acc2);
            int col = nt * 16 + (lane & 15);
            // bias values for the 3 types at this column (registers, no divergent loads)
            float bb0 = bhl[col], bb1 = bhl[64 + col], bb2 = bhl[128 + col];
#pragma unroll
            for (int r = 0; r < 8; ++r) {
                int t = trow[r];
                float av = acc0[r];
                av = (t == 1) ? acc1[r] : av;
                av = (t == 2) ? acc2[r] : av;
                float bb = bb0;
                bb = (t == 1) ? bb1 : bb;
                bb = (t == 2) ? bb2 : bb;
                float v = av + bb;
                int row = row0 + r + hi * 8;
                hout[(size_t)row * 64 + col] = v;
                hstage[wave][r + hi * 8][col] = (_Float16)v;
            }
        }
    }
    __syncthreads();
    if (active) {
        v16h b0 = {}, b1 = {};
        int m = lane & 15;
#pragma unroll
        for (int j = 0; j < 8; ++j) {
            int c0 = a_col(2 * j, hi);
            b0[2 * j]     = hstage[wave][m][c0];
            b0[2 * j + 1] = hstage[wave][m][c0 + 1];
            b1[2 * j]     = hstage[wave][m][32 + c0];
            b1[2 * j + 1] = hstage[wave][m][32 + c0 + 1];
        }
#pragma unroll
        for (int nt = 0; nt < 4; ++nt) {
            v8f c = {};
            c = WMMA16(b0, *(const v16h*)(wsh + 12288 + ((0 * 4 + nt) * 32 + lane) * 16), c);
            c = WMMA16(b1, *(const v16h*)(wsh + 12288 + ((1 * 4 + nt) * 32 + lane) * 16), c);
            int col = nt * 16 + (lane & 15);
#pragma unroll
            for (int r = 0; r < 8; ++r)
                mAout[(size_t)(row0 + r + hi * 8) * 64 + col] = c[r];
        }
    }
}

// ---------------------------------------------------------------------------
// Edge kernel: ea = leaky(e_in @ Wea)  (Kin = 32 or 64), then
// msgB = ea @ WlinB (ea restaged f16 through LDS). One wave = 16 edges;
// packed weights staged once per block into LDS.
// ---------------------------------------------------------------------------
__global__ __launch_bounds__(256) void edge_ea_kernel(
    const float* __restrict__ ein, int Kin, int relu_in,
    const _Float16* __restrict__ pWea,    // (Kin/32) kchunks
    const _Float16* __restrict__ pWlinB,  // 1 mat (K=64)
    float* __restrict__ eaout,            // E x 64
    float* __restrict__ msgB,             // E x 64
    int nTiles) {
    __shared__ _Float16 wshE[8192];        // 16KB: [0,4096) Wea frags, [4096,8192) WlinB
    __shared__ _Float16 estage[8][16][64]; // 16KB
    int KC = Kin >> 5;
    {
        uint4* dsh = (uint4*)wshE;
        const uint4* s0 = (const uint4*)pWea;    // KC*256 x uint4
        int n0 = KC * 256;
        for (int idx = threadIdx.x; idx < n0; idx += 256) dsh[idx] = s0[idx];
        const uint4* s1 = (const uint4*)pWlinB;  // 512 x uint4
        for (int idx = threadIdx.x; idx < 512; idx += 256) dsh[512 + idx] = s1[idx];
    }
    __syncthreads();

    int wave = threadIdx.x >> 5;
    int lane = threadIdx.x & 31;
    int tile = blockIdx.x * 8 + wave;
    bool active = tile < nTiles;
    int row0 = tile * 16;
    int hi   = lane >> 4;
    int mrow = row0 + (lane & 15);
    bool k64 = (Kin == 64);

    if (active) {
        const float* xrow = ein + (size_t)mrow * Kin;
        __builtin_prefetch(ein + (size_t)(mrow + 128) * Kin, 0, 1);
        v16h a0 = {}, a1 = {};
#pragma unroll
        for (int j = 0; j < 8; ++j) {
            int c0 = a_col(2 * j, hi);
            float2 p0 = *(const float2*)(xrow + c0);
            if (relu_in) { p0.x = fmaxf(p0.x, 0.f); p0.y = fmaxf(p0.y, 0.f); }
            a0[2 * j] = (_Float16)p0.x; a0[2 * j + 1] = (_Float16)p0.y;
            if (k64) {
                float2 p1 = *(const float2*)(xrow + 32 + c0);
                if (relu_in) { p1.x = fmaxf(p1.x, 0.f); p1.y = fmaxf(p1.y, 0.f); }
                a1[2 * j] = (_Float16)p1.x; a1[2 * j + 1] = (_Float16)p1.y;
            }
        }
#pragma unroll
        for (int nt = 0; nt < 4; ++nt) {
            v8f c = {};
            c = WMMA16(a0, *(const v16h*)(wshE + ((0 * 4 + nt) * 32 + lane) * 16), c);
            if (k64)
                c = WMMA16(a1, *(const v16h*)(wshE + ((1 * 4 + nt) * 32 + lane) * 16), c);
            int col = nt * 16 + (lane & 15);
#pragma unroll
            for (int r = 0; r < 8; ++r) {
                int row = row0 + r + hi * 8;
                float v = lrelu(c[r]);
                eaout[(size_t)row * 64 + col] = v;
                estage[wave][r + hi * 8][col] = (_Float16)v;
            }
        }
    }
    __syncthreads();
    if (active) {
        v16h b0 = {}, b1 = {};
        int m = lane & 15;
#pragma unroll
        for (int j = 0; j < 8; ++j) {
            int c0 = a_col(2 * j, hi);
            b0[2 * j]     = estage[wave][m][c0];
            b0[2 * j + 1] = estage[wave][m][c0 + 1];
            b1[2 * j]     = estage[wave][m][32 + c0];
            b1[2 * j + 1] = estage[wave][m][32 + c0 + 1];
        }
#pragma unroll
        for (int nt = 0; nt < 4; ++nt) {
            v8f c = {};
            c = WMMA16(b0, *(const v16h*)(wshE + 4096 + ((0 * 4 + nt) * 32 + lane) * 16), c);
            c = WMMA16(b1, *(const v16h*)(wshE + 4096 + ((1 * 4 + nt) * 32 + lane) * 16), c);
            int col = nt * 16 + (lane & 15);
#pragma unroll
            for (int r = 0; r < 8; ++r)
                msgB[(size_t)(row0 + r + hi * 8) * 64 + col] = c[r];
        }
    }
}

// Per-node attention scalars: si = h . Watt[0:64] (dst role), sj = h . Watt[64:128]
__global__ void node_sc_kernel(const float* __restrict__ h, const float* __restrict__ Watt,
                               float* __restrict__ si, float* __restrict__ sj, int N) {
    int n = blockIdx.x * blockDim.x + threadIdx.x;
    if (n >= N) return;
    float a = 0.f, b = 0.f;
#pragma unroll 8
    for (int k = 0; k < 64; ++k) {
        float v = h[(size_t)n * 64 + k];
        a += v * Watt[k];
        b += v * Watt[64 + k];
    }
    si[n] = a; sj[n] = b;
}

__device__ __forceinline__ unsigned f2key(float f) {
    unsigned b = __float_as_uint(f);
    return (b & 0x80000000u) ? ~b : (b | 0x80000000u);
}
__device__ __forceinline__ float key2f(unsigned k) {
    unsigned b = (k & 0x80000000u) ? (k & 0x7fffffffu) : ~k;
    return __uint_as_float(b);
}

// logit = leaky(si[dst] + sj[src] + leaky(Ete[et]).Wt + ea.We); atomic segment max
__global__ void edge_logit_kernel(const int* __restrict__ ei, const int* __restrict__ etype,
                                  const float* __restrict__ ea, const float* __restrict__ Ete,
                                  const float* __restrict__ Watt, const float* __restrict__ si,
                                  const float* __restrict__ sj, float* __restrict__ logit,
                                  unsigned* __restrict__ mkey, int E) {
    int e = blockIdx.x * blockDim.x + threadIdx.x;
    if (e >= E) return;
    int s = ei[e], d = ei[E + e], t = etype[e];
    float ts = 0.f;
#pragma unroll
    for (int k = 0; k < 4; ++k) ts += lrelu(Ete[t * 4 + k]) * Watt[128 + k];
    float se = 0.f;
#pragma unroll 8
    for (int k = 0; k < 64; ++k) se += ea[(size_t)e * 64 + k] * Watt[132 + k];
    float z = lrelu(si[d] + sj[s] + ts + se);
    logit[e] = z;
    atomicMax(&mkey[d], f2key(z));
}

// ex = exp(logit - max[dst]); atomic segment sum
__global__ void edge_exp_kernel(const int* __restrict__ ei, const float* __restrict__ logit,
                                const unsigned* __restrict__ mkey, float* __restrict__ exb,
                                float* __restrict__ ssum, int E) {
    int e = blockIdx.x * blockDim.x + threadIdx.x;
    if (e >= E) return;
    int d = ei[E + e];
    float ex = __expf(logit[e] - key2f(mkey[d]));
    exb[e] = ex;
    atomicAdd(&ssum[d], ex);
}

// out[dst] += (mA[src] + msgB[e]) * alpha; one thread per (edge, channel)
__global__ void edge_msg_kernel(const int* __restrict__ ei, const float* __restrict__ exb,
                                const float* __restrict__ ssum, const float* __restrict__ mA,
                                const float* __restrict__ msgB, float* __restrict__ outb, int E) {
    int tid = blockIdx.x * blockDim.x + threadIdx.x;
    if (tid >= E * 64) return;
    int e = tid >> 6, k = tid & 63;
    int s = ei[e], d = ei[E + e];
    float alpha = exb[e] / (ssum[d] + 1e-16f);
    float v = (mA[(size_t)s * 64 + k] + msgB[(size_t)e * 64 + k]) * alpha;
    atomicAdd(&outb[(size_t)d * 64 + k], v);
}

// x_out[n] = relu(out3[n]) . Wnc[type] + bnc[type]
__global__ void node_out_kernel(const float* __restrict__ hb, const int* __restrict__ ntype,
                                const float* __restrict__ Wnc, const float* __restrict__ bnc,
                                float* __restrict__ out, int N) {
    int n = blockIdx.x * blockDim.x + threadIdx.x;
    if (n >= N) return;
    int t = ntype[n];
    float acc = bnc[t];
#pragma unroll 8
    for (int k = 0; k < 64; ++k)
        acc += fmaxf(hb[(size_t)n * 64 + k], 0.f) * Wnc[t * 64 + k];
    out[n] = acc;
}

// e_out[e] = relu(ea3[e]) . Wec[etype] + bec[etype]
__global__ void edge_out_kernel(const float* __restrict__ eb, const int* __restrict__ etype,
                                const float* __restrict__ Wec, const float* __restrict__ bec,
                                float* __restrict__ out, int E) {
    int e = blockIdx.x * blockDim.x + threadIdx.x;
    if (e >= E) return;
    int t = etype[e];
    float acc = bec[t];
#pragma unroll 8
    for (int k = 0; k < 64; ++k)
        acc += fmaxf(eb[(size_t)e * 64 + k], 0.f) * Wec[t * 64 + k];
    out[e] = acc;
}

extern "C" void kernel_launch(void* const* d_in, const int* in_sizes, int n_in,
                              void* d_out, int out_size, void* d_ws, size_t ws_size,
                              hipStream_t stream) {
    const int N = GNN_N, E = GNN_E;
    const float* x     = (const float*)d_in[0];
    const int*   ei    = (const int*)d_in[1];
    const int*   ntype = (const int*)d_in[2];
    const int*   etype = (const int*)d_in[3];
    const float* eattr = (const float*)d_in[4];
    const float* Whl[3]  = {(const float*)d_in[5],  (const float*)d_in[11], (const float*)d_in[17]};
    const float* bhl[3]  = {(const float*)d_in[6],  (const float*)d_in[12], (const float*)d_in[18]};
    const float* EteP[3] = {(const float*)d_in[7],  (const float*)d_in[13], (const float*)d_in[19]};
    const float* Wea[3]  = {(const float*)d_in[8],  (const float*)d_in[14], (const float*)d_in[20]};
    const float* Watt[3] = {(const float*)d_in[9],  (const float*)d_in[15], (const float*)d_in[21]};
    const float* Wlin[3] = {(const float*)d_in[10], (const float*)d_in[16], (const float*)d_in[22]};
    const float* Wnc = (const float*)d_in[23];
    const float* bnc = (const float*)d_in[24];
    const float* Wec = (const float*)d_in[25];
    const float* bec = (const float*)d_in[26];

    char* w = (char*)d_ws;
    size_t off = 0;
    auto alloc = [&](size_t bytes) -> void* {
        void* p = w + off;
        off = (off + bytes + 255) & ~(size_t)255;
        return p;
    };
    float*     hbuf   = (float*)alloc((size_t)N * 64 * 4);
    float*     mA     = (float*)alloc((size_t)N * 64 * 4);
    float*     nodeO0 = (float*)alloc((size_t)N * 64 * 4);
    float*     nodeO1 = (float*)alloc((size_t)N * 64 * 4);
    float*     si    = (float*)alloc((size_t)N * 4);
    float*     sj    = (float*)alloc((size_t)N * 4);
    float*     ssum  = (float*)alloc((size_t)N * 4);
    unsigned*  mkey  = (unsigned*)alloc((size_t)N * 4);
    float*     logit = (float*)alloc((size_t)E * 4);
    float*     exb   = (float*)alloc((size_t)E * 4);
    float*     eA    = (float*)alloc((size_t)E * 64 * 4);
    float*     eB    = (float*)alloc((size_t)E * 64 * 4);
    float*     msgB  = (float*)alloc((size_t)E * 64 * 4);
    _Float16*  pWhl  = (_Float16*)alloc((size_t)3 * 4096 * 2);
    _Float16*  pWea  = (_Float16*)alloc((size_t)4096 * 2);
    _Float16*  pWlin = (_Float16*)alloc((size_t)2 * 4096 * 2);  // [A | B]

    float* nodeO[2] = {nodeO0, nodeO1};
    const int nTilesN = N / 16;  // 3125
    const int nTilesE = E / 16;  // 50000

    for (int l = 0; l < 3; ++l) {
        int Ke = (l == 0) ? 32 : 64;
        int relu = (l == 0) ? 0 : 1;
        const float* xin  = (l == 0) ? x : nodeO[(l + 1) & 1];
        float*       outb = nodeO[l & 1];
        const float* einp = (l == 0) ? eattr : ((l == 1) ? eA : eB);
        float*       eout = (l == 1) ? eB : eA;

        // pack weights into WMMA B-fragment layout (f16)
        {
            int totHl = 3 * 2 * 2048, totEa = (Ke / 32) * 2048, totLin = 2 * 2 * 2048;
            pack_b_kernel<<<(totHl + 255) / 256, 256, 0, stream>>>(Whl[l], pWhl, 64, 3);
            pack_b_kernel<<<(totEa + 255) / 256, 256, 0, stream>>>(Wea[l], pWea, Ke, 1);
            pack_b_kernel<<<(totLin + 255) / 256, 256, 0, stream>>>(Wlin[l], pWlin, 64, 2);
        }
        hipMemsetAsync(outb, 0, (size_t)N * 64 * 4, stream);
        hipMemsetAsync(mkey, 0, (size_t)N * 4, stream);
        hipMemsetAsync(ssum, 0, (size_t)N * 4, stream);

        node_hl_kernel<<<(nTilesN + 7) / 8, 256, 0, stream>>>(
            xin, relu, ntype, bhl[l], pWhl, pWlin, hbuf, mA, nTilesN);
        node_sc_kernel<<<(N + 255) / 256, 256, 0, stream>>>(hbuf, Watt[l], si, sj, N);
        edge_ea_kernel<<<(nTilesE + 7) / 8, 256, 0, stream>>>(
            einp, Ke, relu, pWea, pWlin + 4096, eout, msgB, nTilesE);
        edge_logit_kernel<<<(E + 255) / 256, 256, 0, stream>>>(
            ei, etype, eout, EteP[l], Watt[l], si, sj, logit, mkey, E);
        edge_exp_kernel<<<(E + 255) / 256, 256, 0, stream>>>(ei, logit, mkey, exb, ssum, E);
        edge_msg_kernel<<<(E * 64 + 255) / 256, 256, 0, stream>>>(
            ei, exb, ssum, mA, msgB, outb, E);
    }

    // heads: layer-3 node output is in nodeO[0], layer-3 ea is in eA
    node_out_kernel<<<(N + 255) / 256, 256, 0, stream>>>(
        nodeO[0], ntype, Wnc, bnc, (float*)d_out, N);
    edge_out_kernel<<<(E + 255) / 256, 256, 0, stream>>>(
        eA, etype, Wec, bec, (float*)d_out + N, E);
}
